// CausalSelfAttention_38457137169022
// MI455X (gfx1250) — compile-verified, matrix operand order checked by reading, not run on hardware
//
#include <hip/hip_runtime.h>

// ---------------------------------------------------------------------------
// Causal self-attention for MI455X (gfx1250, wave32, WMMA).
// fp32 reference emulated with bf16x3 split-precision WMMA (fp32 accumulate):
//   A*B ~= Ahi*Bhi + Ahi*Blo + Alo*Bhi   (error ~2^-16, fp32-class)
// Pipeline:
//   [QKV gemm -> pre-split bf16 hi/lo planes, V transposed, Q pre-scaled]
//   -> [flash attention: Q via global_load_async_to_lds_b128 (ASYNCcnt),
//       K/V tiles via TDM tensor_load_to_lds (TENSORcnt), double-buffered;
//       softmax row-sums computed on the WMMA pipe as P @ Ones]
//   -> [out gemm]
// Fragments load from LDS as 2x ds_load_b128 (stride % 8 == 0).
// ---------------------------------------------------------------------------

#define EMBED   1024
#define NHEADS  16
#define HDIM    64
#define BATCH   2
#define SEQ     2048
#define MROWS   (BATCH * SEQ)   // 4096

typedef __attribute__((ext_vector_type(16))) __bf16 v16bf;
typedef __attribute__((ext_vector_type(2)))  __bf16 v2bf;
typedef __attribute__((ext_vector_type(8)))  float  v8f;
typedef __attribute__((ext_vector_type(4)))  unsigned v4u;
typedef __attribute__((ext_vector_type(8)))  unsigned v8u;

union AccF { v8f v; float f[8]; };

// ---- fp32 -> bf16 (RNE) helpers -------------------------------------------
__device__ __forceinline__ unsigned short f2bf_rne(float x) {
    unsigned u = __builtin_bit_cast(unsigned, x);
    unsigned r = u + 0x7FFFu + ((u >> 16) & 1u);
    return (unsigned short)(r >> 16);
}
__device__ __forceinline__ float bf2f(unsigned short h) {
    unsigned u = ((unsigned)h) << 16;
    return __builtin_bit_cast(float, u);
}
// Two fp32 -> packed bf16 pair (RNE). Uses v_cvt_pk_bf16_f32 when available.
__device__ __forceinline__ unsigned cvt_pk_bf16(float a, float b) {
#if __has_builtin(__builtin_amdgcn_cvt_pk_bf16_f32)
    v2bf r = __builtin_amdgcn_cvt_pk_bf16_f32(a, b);
    return __builtin_bit_cast(unsigned, r);
#else
    return (unsigned)f2bf_rne(a) | ((unsigned)f2bf_rne(b) << 16);
#endif
}
// Split pair (a,b) into packed bf16 hi parts and packed bf16 residuals.
__device__ __forceinline__ void split2(float a, float b, unsigned& hi, unsigned& lo) {
    hi = cvt_pk_bf16(a, b);
    float ha = bf2f((unsigned short)hi);
    float hb = bf2f((unsigned short)(hi >> 16));
    lo = cvt_pk_bf16(a - ha, b - hb);
}

// ---- fragment load from LDS: two contiguous 16B chunks per lane -----------
// 16-bit A/B operand (16x32): lanes 0-15 K=kbase+{0..7,16..23}, lanes 16-31 +8.
template <int STRIDE>
__device__ __forceinline__ v16bf load_frag(const unsigned short* lds, int row,
                                           int kbase, int laneHiOff) {
    union { v16bf v; uint4 q[2]; } f;
    const unsigned short* p = lds + row * STRIDE + kbase + laneHiOff;
    f.q[0] = *(const uint4*)(p);
    f.q[1] = *(const uint4*)(p + 16);
    return f.v;
}

// ---- bf16x3 split-precision MMA -------------------------------------------
__device__ __forceinline__ v8f mma_x3(v8f acc, v16bf ah, v16bf al, v16bf bh, v16bf bl) {
    acc = __builtin_amdgcn_wmma_f32_16x16x32_bf16(false, ah, false, bh, (short)0, acc, false, false);
    acc = __builtin_amdgcn_wmma_f32_16x16x32_bf16(false, ah, false, bl, (short)0, acc, false, false);
    acc = __builtin_amdgcn_wmma_f32_16x16x32_bf16(false, al, false, bh, (short)0, acc, false, false);
    return acc;
}
__device__ __forceinline__ v8f mma1(v8f acc, v16bf a, v16bf b) {
    return __builtin_amdgcn_wmma_f32_16x16x32_bf16(false, a, false, b, (short)0, acc, false, false);
}

// ---- CDNA5 async global->LDS (ASYNCcnt) -----------------------------------
__device__ __forceinline__ unsigned lds_addr_u32(const void* p) {
    return (unsigned)(size_t)p;   // LDS aperture: addr[31:0] is the LDS byte offset
}
__device__ __forceinline__ void async_b128(unsigned lds, const void* gbase, unsigned voff) {
    asm volatile("global_load_async_to_lds_b128 %0, %1, %2"
                 :: "v"(lds), "v"(voff), "s"(gbase)
                 : "memory");
}
template <int N>
__device__ __forceinline__ void wait_async() {
    asm volatile("s_wait_asynccnt %0" :: "i"(N) : "memory");
}
// Copy a 64-row x 128-byte tile into LDS (row stride 144B). 128 threads.
__device__ __forceinline__ void async_copy_tile(const unsigned short* gbase,
                                                int rowStrideBytes,
                                                unsigned ldsBase, int tid) {
#pragma unroll
    for (int i = 0; i < 4; ++i) {
        int c = tid + i * 128;
        int row = c >> 3;
        int off = (c & 7) << 4;
        async_b128(ldsBase + (unsigned)(row * 144 + off), gbase,
                   (unsigned)(row * rowStrideBytes + off));
    }
}

// ---- CDNA5 TDM: one-shot 64x64-element (bf16) tile -> LDS (TENSORcnt) -----
// 2D descriptor (groups 2/3 NULL). data_size=2B; dims/strides in elements.
// LDS pad: 4 dwords after every 32 dwords -> 144B LDS row stride (ST=72).
__device__ __forceinline__ void tdm_load_tile(const void* gaddr, unsigned ldsByteAddr,
                                              unsigned rowStrideElems) {
    unsigned long long ga = (unsigned long long)(size_t)gaddr;
    v4u g0;
    g0[0] = 1u;                                   // count=1, user descriptor
    g0[1] = ldsByteAddr;                          // lds_addr[31:0]
    g0[2] = (unsigned)ga;                         // global_addr[31:0]
    g0[3] = (unsigned)(ga >> 32) | (2u << 30);    // global_addr[56:32] | type=2
    v8u g1;
    g1[0] = (1u << 16) | (1u << 20) | (4u << 22) | (3u << 25);
    //      data_size=2B | pad_enable | pad_interval=32dw | pad_amount=4dw
    g1[1] = (64u << 16);                          // tensor_dim0 = 64
    g1[2] = (64u << 16);                          // tensor_dim1 = 64 (low16)
    g1[3] = (64u << 16);                          // tile_dim0 = 64
    g1[4] = 64u;                                  // tile_dim1 = 64, tile_dim2 = 0
    g1[5] = rowStrideElems;                       // tensor_dim0_stride[31:0]
    g1[6] = 0u;
    g1[7] = 0u;
    asm volatile("tensor_load_to_lds %0, %1" :: "s"(g0), "s"(g1) : "memory");
}

// ---------------------------------------------------------------------------
// GEMM: C[M,N] = A[M,K] * B[K,N] (both fp32). Block tile 128x128, K-slice 32.
// ---------------------------------------------------------------------------
struct QkvStore {   // scatter into pre-split bf16 planes; V transposed [b][h][d][t]
    unsigned short *qh, *ql, *kh, *kl, *vth, *vtl;
    // 8 consecutive rows (gm0..gm0+7) at one column gn, per lane.
    __device__ __forceinline__ void store8(int gm0, int gn, const float* vals) const {
        int which = gn >> 10;                 // uniform per block (128 | 1024)
        int h = (gn >> 6) & (NHEADS - 1);
        int d = gn & (HDIM - 1);
        int b = gm0 >> 11, t0 = gm0 & (SEQ - 1);
        int bh = b * NHEADS + h;
        if (which == 2) {                     // V: 8 consecutive t -> packed 16B
            unsigned hv[4], lv[4];
#pragma unroll
            for (int r = 0; r < 8; r += 2)
                split2(vals[r], vals[r + 1], hv[r >> 1], lv[r >> 1]);
            size_t i = ((size_t)bh * HDIM + d) * SEQ + t0;
            *(uint4*)(vth + i) = make_uint4(hv[0], hv[1], hv[2], hv[3]);
            *(uint4*)(vtl + i) = make_uint4(lv[0], lv[1], lv[2], lv[3]);
        } else {
            // Q gets the attention scale folded in (exact: power of two).
            const float sc = (which == 0) ? 0.125f : 1.0f;
            unsigned short* ph = (which == 0) ? qh : kh;
            unsigned short* pl = (which == 0) ? ql : kl;
            size_t i0 = ((size_t)bh * SEQ + t0) * HDIM + d;
#pragma unroll
            for (int r = 0; r < 8; r += 2) {
                unsigned hi2, lo2;
                split2(vals[r] * sc, vals[r + 1] * sc, hi2, lo2);
                ph[i0 + (size_t)r * HDIM]       = (unsigned short)hi2;
                ph[i0 + (size_t)(r + 1) * HDIM] = (unsigned short)(hi2 >> 16);
                pl[i0 + (size_t)r * HDIM]       = (unsigned short)lo2;
                pl[i0 + (size_t)(r + 1) * HDIM] = (unsigned short)(lo2 >> 16);
            }
        }
    }
};
struct PlainStore {
    float* out; int ldn;
    __device__ __forceinline__ void store8(int gm0, int gn, const float* vals) const {
#pragma unroll
        for (int r = 0; r < 8; ++r)
            out[(size_t)(gm0 + r) * ldn + gn] = vals[r];
    }
};

template <class StoreOp>
__global__ __launch_bounds__(256) void gemm_bf16x3_kernel(
    const float* __restrict__ A, const float* __restrict__ Bw,
    int M, int N, int K, StoreOp op) {
    constexpr int AST = 40;  // 32 + pad(8): 20-dword stride, conflict-free, 16B aligned
    __shared__ __align__(16) unsigned short Ahi[128 * AST], Alo[128 * AST];
    __shared__ __align__(16) unsigned short Bhi[128 * AST], Blo[128 * AST];  // n-major

    const int tid  = threadIdx.x;
    const int lane = tid & 31, wid = tid >> 5;
    const int mBase = blockIdx.y * 128;
    const int nBase = blockIdx.x * 128;
    const int waveM = (wid & 1) * 64;
    const int waveN = (wid >> 1) * 32;
    const int laneHiOff = (lane & 16) ? 8 : 0;
    const int lrow  = lane & 15;
    const int mHalf = (lane < 16) ? 0 : 8;

    AccF acc[4][2];
#pragma unroll
    for (int mt = 0; mt < 4; ++mt)
#pragma unroll
        for (int nt = 0; nt < 2; ++nt)
#pragma unroll
            for (int r = 0; r < 8; ++r) acc[mt][nt].f[r] = 0.f;

    const int bn = tid & 127;
    const int bkq = (tid >> 7) * 4;

    for (int kb = 0; kb < K; kb += 32) {
#pragma unroll
        for (int i = 0; i < 4; ++i) {
            int linear = tid + i * 256;
            int row = linear >> 3;
            int col = (linear & 7) << 2;
            float4 x = *(const float4*)(A + (size_t)(mBase + row) * K + kb + col);
            unsigned h01, l01, h23, l23;
            split2(x.x, x.y, h01, l01);
            split2(x.z, x.w, h23, l23);
            *(uint2*)&Ahi[row * AST + col] = make_uint2(h01, h23);
            *(uint2*)&Alo[row * AST + col] = make_uint2(l01, l23);
        }
#pragma unroll
        for (int g = 0; g < 4; ++g) {
            int k0 = bkq + g * 8;
            float x0 = Bw[(size_t)(kb + k0 + 0) * N + nBase + bn];
            float x1 = Bw[(size_t)(kb + k0 + 1) * N + nBase + bn];
            float x2 = Bw[(size_t)(kb + k0 + 2) * N + nBase + bn];
            float x3 = Bw[(size_t)(kb + k0 + 3) * N + nBase + bn];
            unsigned h01, l01, h23, l23;
            split2(x0, x1, h01, l01);
            split2(x2, x3, h23, l23);
            *(uint2*)&Bhi[bn * AST + k0] = make_uint2(h01, h23);
            *(uint2*)&Blo[bn * AST + k0] = make_uint2(l01, l23);
        }
        __syncthreads();

        v16bf ah[4], al[4];
#pragma unroll
        for (int mt = 0; mt < 4; ++mt) {
            int r = waveM + mt * 16 + lrow;
            ah[mt] = load_frag<AST>(Ahi, r, 0, laneHiOff);
            al[mt] = load_frag<AST>(Alo, r, 0, laneHiOff);
        }
#pragma unroll
        for (int nt = 0; nt < 2; ++nt) {
            int r = waveN + nt * 16 + lrow;
            v16bf bh = load_frag<AST>(Bhi, r, 0, laneHiOff);
            v16bf bl = load_frag<AST>(Blo, r, 0, laneHiOff);
#pragma unroll
            for (int mt = 0; mt < 4; ++mt)
                acc[mt][nt].v = mma_x3(acc[mt][nt].v, ah[mt], al[mt], bh, bl);
        }
        __syncthreads();
    }

#pragma unroll
    for (int mt = 0; mt < 4; ++mt)
#pragma unroll
        for (int nt = 0; nt < 2; ++nt)
            op.store8(mBase + waveM + mt * 16 + mHalf,
                      nBase + waveN + nt * 16 + lrow, acc[mt][nt].f);
}

// ---------------------------------------------------------------------------
// Flash attention: one block per (b, h, 64-row q tile). 128 threads = 4 waves.
// Q staged with async global->LDS; K/V tiles staged by the TDM (wave 0 issues
// one tensor_load_to_lds per plane), double-buffered across the kt loop.
// Row-sums of P computed as P @ Ones on the WMMA pipe (replicated per lane).
// ---------------------------------------------------------------------------
__global__ __launch_bounds__(128) void flash_attn_kernel(
    const unsigned short* __restrict__ Qh, const unsigned short* __restrict__ Ql,
    const unsigned short* __restrict__ Kh, const unsigned short* __restrict__ Kl,
    const unsigned short* __restrict__ Vth, const unsigned short* __restrict__ Vtl,
    float* __restrict__ Og) {
    constexpr int ST = 72;  // 64 + pad(8): matches TDM pad (144B LDS rows)
    __shared__ __align__(16) unsigned short Qhi[64 * ST], Qlo[64 * ST];
    __shared__ __align__(16) unsigned short KV[2][4][64 * ST];  // [buf][Khi,Klo,Vthi,Vtlo]
    __shared__ __align__(16) unsigned short Phi[4][16 * ST], Plo[4][16 * ST];

    const int tid  = threadIdx.x;
    const int lane = tid & 31, wid = tid >> 5;
    const int qt = blockIdx.x & 31;
    const int bh = blockIdx.x >> 5;
    const int b = bh >> 4, h = bh & 15;

    const unsigned short* qhp  = Qh  + ((size_t)bh * SEQ + qt * 64) * HDIM;
    const unsigned short* qlp  = Ql  + ((size_t)bh * SEQ + qt * 64) * HDIM;
    const unsigned short* khp  = Kh  + (size_t)bh * SEQ * HDIM;
    const unsigned short* klp  = Kl  + (size_t)bh * SEQ * HDIM;
    const unsigned short* vthp = Vth + (size_t)bh * HDIM * SEQ;
    const unsigned short* vtlp = Vtl + (size_t)bh * HDIM * SEQ;

    const int laneHiOff = (lane & 16) ? 8 : 0;
    const int lrow  = lane & 15;
    const int mHalf = (lane < 16) ? 0 : 8;

    // Constant bf16 all-ones B operand for WMMA row-sum (P @ Ones).
    union { v16bf v; unsigned u[8]; } onesU;
#pragma unroll
    for (int j = 0; j < 8; ++j) onesU.u[j] = 0x3F803F80u;
    const v16bf ones = onesU.v;

    auto stage_kv_tdm = [&](int kt2, int buf) {   // wave 0 only
        const size_t koff = (size_t)kt2 * 64 * HDIM;  // K rows contiguous
        tdm_load_tile(khp + koff, lds_addr_u32(&KV[buf][0][0]), HDIM);
        tdm_load_tile(klp + koff, lds_addr_u32(&KV[buf][1][0]), HDIM);
        tdm_load_tile(vthp + kt2 * 64, lds_addr_u32(&KV[buf][2][0]), SEQ);
        tdm_load_tile(vtlp + kt2 * 64, lds_addr_u32(&KV[buf][3][0]), SEQ);
    };

    // Prologue: Q via async (all waves), first K/V tile via TDM (wave 0).
    async_copy_tile(qhp, 128, lds_addr_u32(Qhi), tid);
    async_copy_tile(qlp, 128, lds_addr_u32(Qlo), tid);
    if (wid == 0) stage_kv_tdm(0, 0);
    wait_async<0>();
    __syncthreads();

    v16bf qh[2], ql[2];
#pragma unroll
    for (int ks = 0; ks < 2; ++ks) {
        qh[ks] = load_frag<ST>(Qhi, wid * 16 + lrow, ks * 32, laneHiOff);
        ql[ks] = load_frag<ST>(Qlo, wid * 16 + lrow, ks * 32, laneHiOff);
    }

    AccF oacc[4];
    float m_run[8], l_run[8];
#pragma unroll
    for (int nt = 0; nt < 4; ++nt)
#pragma unroll
        for (int r = 0; r < 8; ++r) oacc[nt].f[r] = 0.f;
#pragma unroll
    for (int r = 0; r < 8; ++r) { m_run[r] = -3.0e38f; l_run[r] = 0.f; }

    const int qrow0 = qt * 64 + wid * 16 + mHalf;

    for (int kt = 0; kt <= qt; ++kt) {
        const int cur = kt & 1;
        if (wid == 0) {
            if (kt < qt) {           // overlap next tile's DMA with this tile's math
                stage_kv_tdm(kt + 1, cur ^ 1);
                __builtin_amdgcn_s_wait_tensorcnt(4);  // current tile landed
            } else {
                __builtin_amdgcn_s_wait_tensorcnt(0);
            }
        }
        __syncthreads();

        const unsigned short* KhiL  = &KV[cur][0][0];
        const unsigned short* KloL  = &KV[cur][1][0];
        const unsigned short* VthiL = &KV[cur][2][0];
        const unsigned short* VtloL = &KV[cur][3][0];

        // S = Q * K^T (16 q rows x 64 keys per wave; scale pre-folded into Q).
        AccF sacc[4];
#pragma unroll
        for (int nt = 0; nt < 4; ++nt) {
#pragma unroll
            for (int r = 0; r < 8; ++r) sacc[nt].f[r] = 0.f;
#pragma unroll
            for (int ks = 0; ks < 2; ++ks) {
                v16bf kbh = load_frag<ST>(KhiL, nt * 16 + lrow, ks * 32, laneHiOff);
                v16bf kbl = load_frag<ST>(KloL, nt * 16 + lrow, ks * 32, laneHiOff);
                sacc[nt].v = mma_x3(sacc[nt].v, qh[ks], ql[ks], kbh, kbl);
            }
        }
        // Causal mask: only the diagonal tile can mask (kt < qt is fully valid).
        if (kt == qt) {
#pragma unroll
            for (int nt = 0; nt < 4; ++nt) {
                int kcol = kt * 64 + nt * 16 + lrow;
#pragma unroll
                for (int r = 0; r < 8; ++r)
                    if (kcol > qrow0 + r) sacc[nt].f[r] = -3.0e38f;
            }
        }
        // Online softmax: row max across lanes 0-15 / 16-31 halves; exponentials.
        float alpha[8];
#pragma unroll
        for (int r = 0; r < 8; ++r) {
            float rm = fmaxf(fmaxf(sacc[0].f[r], sacc[1].f[r]),
                             fmaxf(sacc[2].f[r], sacc[3].f[r]));
            rm = fmaxf(rm, __shfl_xor(rm, 1));
            rm = fmaxf(rm, __shfl_xor(rm, 2));
            rm = fmaxf(rm, __shfl_xor(rm, 4));
            rm = fmaxf(rm, __shfl_xor(rm, 8));
            float mn = fmaxf(m_run[r], rm);
            alpha[r] = __expf(m_run[r] - mn);
            m_run[r] = mn;
#pragma unroll
            for (int nt = 0; nt < 4; ++nt)
                sacc[nt].f[r] = __expf(sacc[nt].f[r] - mn);
#pragma unroll
            for (int d4 = 0; d4 < 4; ++d4) oacc[d4].f[r] *= alpha[r];
        }
        // P -> per-wave LDS (hi/lo) to re-enter in A-fragment layout.
        unsigned short* ph = &Phi[wid][0];
        unsigned short* pl = &Plo[wid][0];
#pragma unroll
        for (int nt = 0; nt < 4; ++nt)
#pragma unroll
            for (int r = 0; r < 8; r += 2) {
                int col = nt * 16 + lrow;
                unsigned hi2, lo2;
                split2(sacc[nt].f[r], sacc[nt].f[r + 1], hi2, lo2);
                ph[(mHalf + r) * ST + col]     = (unsigned short)hi2;
                ph[(mHalf + r + 1) * ST + col] = (unsigned short)(hi2 >> 16);
                pl[(mHalf + r) * ST + col]     = (unsigned short)lo2;
                pl[(mHalf + r + 1) * ST + col] = (unsigned short)(lo2 >> 16);
            }
        // O += P * V, and row-sums on the WMMA pipe: racc = P @ Ones
        // (every output column equals the row sum -> replicated per lane).
        AccF racc;
#pragma unroll
        for (int r = 0; r < 8; ++r) racc.f[r] = 0.f;
#pragma unroll
        for (int ks = 0; ks < 2; ++ks) {
            v16bf pah = load_frag<ST>(ph, lrow, ks * 32, laneHiOff);
            v16bf pal = load_frag<ST>(pl, lrow, ks * 32, laneHiOff);
            racc.v = mma1(racc.v, pah, ones);
            racc.v = mma1(racc.v, pal, ones);
#pragma unroll
            for (int nt = 0; nt < 4; ++nt) {
                v16bf vbh = load_frag<ST>(VthiL, nt * 16 + lrow, ks * 32, laneHiOff);
                v16bf vbl = load_frag<ST>(VtloL, nt * 16 + lrow, ks * 32, laneHiOff);
                oacc[nt].v = mma_x3(oacc[nt].v, pah, pal, vbh, vbl);
            }
        }
#pragma unroll
        for (int r = 0; r < 8; ++r)
            l_run[r] = l_run[r] * alpha[r] + racc.f[r];
        __syncthreads();   // readers done before next iteration's DMA overwrites
    }

    // Epilogue: normalize and write per-head output back to [B*T, EMBED].
#pragma unroll
    for (int nt = 0; nt < 4; ++nt)
#pragma unroll
        for (int r = 0; r < 8; ++r) {
            int trow = qt * 64 + wid * 16 + mHalf + r;
            int col  = h * HDIM + nt * 16 + lrow;
            Og[((size_t)(b * SEQ + trow)) * EMBED + col] = oacc[nt].f[r] / l_run[r];
        }
}

// ---------------------------------------------------------------------------
extern "C" void kernel_launch(void* const* d_in, const int* in_sizes, int n_in,
                              void* d_out, int out_size, void* d_ws, size_t ws_size,
                              hipStream_t stream) {
    (void)in_sizes; (void)n_in; (void)out_size; (void)ws_size;
    const float* x     = (const float*)d_in[0];
    const float* w_qkv = (const float*)d_in[1];
    const float* w_out = (const float*)d_in[2];
    float* out = (float*)d_out;

    const size_t perMat = (size_t)BATCH * NHEADS * SEQ * HDIM;  // 4,194,304 elems
    unsigned short* qh  = (unsigned short*)d_ws;
    unsigned short* ql  = qh  + perMat;
    unsigned short* kh  = ql  + perMat;
    unsigned short* kl  = kh  + perMat;
    unsigned short* vth = kl  + perMat;
    unsigned short* vtl = vth + perMat;
    float* o_ws = (float*)(vtl + perMat);                        // [4096][1024] fp32

    QkvStore qs{qh, ql, kh, kl, vth, vtl};
    gemm_bf16x3_kernel<QkvStore>
        <<<dim3(3 * EMBED / 128, MROWS / 128), 256, 0, stream>>>(
            x, w_qkv, MROWS, 3 * EMBED, EMBED, qs);

    flash_attn_kernel<<<dim3(BATCH * NHEADS * (SEQ / 64)), 128, 0, stream>>>(
        qh, ql, kh, kl, vth, vtl, o_ws);

    PlainStore ps{out, EMBED};
    gemm_bf16x3_kernel<PlainStore>
        <<<dim3(EMBED / 128, MROWS / 128), 256, 0, stream>>>(
            o_ws, w_out, MROWS, EMBED, EMBED, ps);
}
